// HighlightDetectionPredictionHead_90237262889049
// MI455X (gfx1250) — compile-verified, matrix operand order checked by reading, not run on hardware
//
#include <hip/hip_runtime.h>
#include <hip/hip_bf16.h>
#include <math.h>

// ---------------------------------------------------------------------------
// HighlightDetectionPredictionHead for MI455X (gfx1250, wave32, WMMA + async)
//   B=64, NQ=10, T=2048, D=256, gates G=768
// ---------------------------------------------------------------------------

typedef __attribute__((ext_vector_type(16))) __bf16 v16bf;
typedef __attribute__((ext_vector_type(8)))  float  v8f;

#define BQ   64
#define NQ   10
#define TT   2048
#define DD   256
#define GG   768
#define NTILES 48          // 768/16
#define KTILES 8           // 256/32
#define PACK_ELEMS (GG*DD) // 196608 bf16 per matrix

// ---------------------------------------------------------------------------
// helpers
// ---------------------------------------------------------------------------
__device__ __forceinline__ uint32_t pack2bf(float a, float b) {
    const uint16_t ua = __builtin_bit_cast(uint16_t, (__bf16)a);
    const uint16_t ub = __builtin_bit_cast(uint16_t, (__bf16)b);
    return (uint32_t)ua | ((uint32_t)ub << 16);
}

// 16 floats -> two contiguous 8x bf16 runs (one b128 store each)
__device__ __forceinline__ void pack_store_frag(__bf16* lo, __bf16* hi, const float* v) {
    uint32_t p[8];
    #pragma unroll
    for (int i = 0; i < 8; ++i) p[i] = pack2bf(v[2*i], v[2*i + 1]);
    *(uint4*)lo = make_uint4(p[0], p[1], p[2], p[3]);
    *(uint4*)hi = make_uint4(p[4], p[5], p[6], p[7]);
}

// async copy 64B global -> LDS (4x GLOBAL_LOAD_ASYNC_TO_LDS_B128; the 24-bit
// instruction offset is added to BOTH the LDS and global addresses, ISA 10.x)
__device__ __forceinline__ void async_copy64(uint32_t lds_off, const void* gaddr) {
    asm volatile(
        "global_load_async_to_lds_b128 %0, %1, off\n\t"
        "global_load_async_to_lds_b128 %0, %1, off offset:16\n\t"
        "global_load_async_to_lds_b128 %0, %1, off offset:32\n\t"
        "global_load_async_to_lds_b128 %0, %1, off offset:48"
        :: "v"(lds_off), "v"(gaddr) : "memory");
}

__device__ __forceinline__ void wait_async0() {
#if __has_builtin(__builtin_amdgcn_s_wait_asynccnt)
    __builtin_amdgcn_s_wait_asynccnt(0);
#else
    asm volatile("s_wait_asynccnt 0x0" ::: "memory");
#endif
}

// ---------------------------------------------------------------------------
// Kernel 1: span selection + collapsed-projection vector + bias sum
// ---------------------------------------------------------------------------
__global__ __launch_bounds__(256) void prep_kernel(
    const float* __restrict__ logits,   // (64,10,2)
    const float* __restrict__ spans,    // (64,10,2)
    const float* __restrict__ W_sal,    // (256,256)
    const float* __restrict__ b_sal,    // (256)
    int* __restrict__ start, int* __restrict__ endv,
    float* __restrict__ wcol, float* __restrict__ bsum)
{
    const int tid = threadIdx.x;
    if (tid < BQ) {
        const int b = tid;
        float best = -1e30f; int bq = 0;
        for (int q = 0; q < NQ; ++q) {
            // softmax(...)[0] is monotone in (l0 - l1)
            float s = logits[(b*NQ + q)*2 + 0] - logits[(b*NQ + q)*2 + 1];
            if (s > best) { best = s; bq = q; }
        }
        const float cx = spans[(b*NQ + bq)*2 + 0];
        const float w  = spans[(b*NQ + bq)*2 + 1];
        // xx = val * (T*CLIP_LEN); se = floor(xx / CLIP_LEN) = floor(val * T)
        const int se0 = (int)floorf((cx - 0.5f*w) * (float)TT);
        const int se1 = (int)floorf((cx + 0.5f*w) * (float)TT);
        start[b] = min(max(se0, 0), TT - 1);
        endv[b]  = min(se1, TT - 1);
    }
    {   // wcol[d] = sum_e W_sal[e,d]
        const int d = tid;
        float s = 0.f;
        for (int e = 0; e < DD; ++e) s += W_sal[e*DD + d];
        wcol[d] = s;
    }
    if (tid == 0) {
        float bs = 0.f;
        for (int e = 0; e < DD; ++e) bs += b_sal[e];
        *bsum = bs;
    }
}

// ---------------------------------------------------------------------------
// Kernel 2: pack W_ih / W_hh (fp32, 768x256 row-major, gi = x . W^T)
// into bf16 WMMA B-fragment order:
//   pack[((nt*8+kt)*32 + lane)*16 + e] = W[g, K]
//   g = nt*16 + (lane&15)
//   K = kt*32 + (e>=8 ? 16:0) + (lane>>4)*8 + (e&7)
// ---------------------------------------------------------------------------
__global__ __launch_bounds__(256) void pack_kernel(
    const float* __restrict__ W_ih, const float* __restrict__ W_hh,
    __bf16* __restrict__ packIH, __bf16* __restrict__ packHH)
{
    const int gid = blockIdx.x * 256 + threadIdx.x;      // 0 .. 2*196608-1
    const float* W = (gid < PACK_ELEMS) ? W_ih : W_hh;
    __bf16*      P = (gid < PACK_ELEMS) ? packIH : packHH;
    const int r    = (gid < PACK_ELEMS) ? gid : gid - PACK_ELEMS;
    const int e    = r & 15;
    const int lane = (r >> 4) & 31;
    const int kt   = (r >> 9) & 7;
    const int nt   = r >> 12;
    const int g    = nt * 16 + (lane & 15);
    const int K    = kt * 32 + ((e >= 8) ? 16 : 0) + (lane >> 4) * 8 + (e & 7);
    P[r] = (__bf16)W[g * DD + K];
}

// ---------------------------------------------------------------------------
// Kernel 3: fused GRU scan. 4 blocks x 256 threads (8 waves, wave32).
// Per step: WMMA phase reads bf16 A-fragments (h and x_t) from LDS and
// L2-resident packed B-fragments; meanwhile async-to-LDS staging pulls the
// x_{t+1} rows (double buffered). The elementwise gate phase updates h and
// converts the staged fp32 row to bf16 fragments for the next step.
// 96 v_wmma_f32_16x16x32_bf16 per wave per step.
// ---------------------------------------------------------------------------
__global__ __launch_bounds__(256, 1) void gru_kernel(
    const float* __restrict__ src_vid,   // (64,2048,256)
    const float* __restrict__ b_ih,      // (768)
    const float* __restrict__ b_hh,      // (768)
    const int*   __restrict__ start,
    const int*   __restrict__ endv,
    const __bf16* __restrict__ packIH,
    const __bf16* __restrict__ packHH,
    float* __restrict__ hidden)          // (64,256)
{
    extern __shared__ __align__(32) char smem[];
    float (*g_rz)[512]  = (float(*)[512]) (smem);            // 32 KB  cols 0..511
    float (*g_in)[256]  = (float(*)[256]) (smem + 32768);    // 16 KB  n-gate x part
    float (*g_hn)[256]  = (float(*)[256]) (smem + 49152);    // 16 KB  n-gate h part
    float (*h_row)[256] = (float(*)[256]) (smem + 65536);    // 16 KB  h fp32 row-major
    __bf16* h_frag      = (__bf16*)       (smem + 81920);    //  8 KB  [k][lane][e]
    __bf16* x_frag      = (__bf16*)       (smem + 90112);    // 16 KB  [2][k][lane][e]
    float*  x_raw       = (float*)        (smem + 106496);   // 32 KB  [2][16][256]

    const int tid  = threadIdx.x;
    const int lane = tid & 31;
    const int wave = tid >> 5;
    const int m    = lane & 15;       // WMMA C/D column (N) for this lane
    const int hlf  = lane >> 4;       // lane half -> C/D row offset

    // elementwise / staging mapping: thread -> (row mm, 16 consecutive cols)
    const int mm = tid >> 4;
    const int j0 = (tid & 15) * 16;
    const int bb = blockIdx.x * 16 + mm;
    const int start_mm = start[bb];
    const int end_mm   = endv[bb];
    const int kT = j0 >> 5;                      // K-tile of this 16-col run
    const int e0 = ((j0 >> 4) & 1) * 8;          // fragment element offset
    __bf16* hfrag_lo = h_frag + (kT*32 + mm)*16 + e0;
    __bf16* hfrag_hi = h_frag + (kT*32 + mm + 16)*16 + e0;
    const int xf_lo = (kT*32 + mm)*16 + e0;      // offsets within one x_frag buf
    const int xf_hi = (kT*32 + mm + 16)*16 + e0;

    // per-wave bias registers (hoisted out of the scan)
    const int rz0 = wave * 4;         // global N-tiles 0..31  (r and z gates)
    const int nn0 = 32 + wave * 2;    // global N-tiles 32..47 (n gate)
    float bias_rz[4], bias_in[2], bias_hn[2];
    #pragma unroll
    for (int i = 0; i < 4; ++i)
        bias_rz[i] = b_ih[(rz0 + i)*16 + m] + b_hh[(rz0 + i)*16 + m];
    #pragma unroll
    for (int i = 0; i < 2; ++i) {
        bias_in[i] = b_ih[(nn0 + i)*16 + m];
        bias_hn[i] = b_hh[(nn0 + i)*16 + m];
    }

    // zero initial hidden state
    for (int i = tid; i < 16*256; i += 256) h_row[i >> 8][i & 255] = 0.f;
    for (int i = tid; i < 8*32*16/2; i += 256) ((uint32_t*)h_frag)[i] = 0u;

    // prologue: stage x_0 fragments directly from global
    {
        const float* gx = src_vid + ((size_t)bb*TT + start_mm)*DD + j0;
        const float msk = (start_mm <= end_mm) ? 1.f : 0.f;
        float xv[16];
        #pragma unroll
        for (int jj = 0; jj < 16; ++jj) xv[jj] = gx[jj] * msk;
        pack_store_frag(x_frag + xf_lo, x_frag + xf_hi, xv);
    }
    __syncthreads();

    for (int t = 0; t < TT; ++t) {
        const int cur = t & 1, nxt = cur ^ 1;

        // ---- async stage x_{t+1} raw rows into LDS (overlaps the WMMAs) ----
        if (t + 1 < TT) {
            int rn = start_mm + t + 1; rn = rn < (TT - 1) ? rn : (TT - 1);
            const int chunk = tid & 15;             // 64B float granules x16
            const float* ga = src_vid + ((size_t)bb*TT + rn)*DD + chunk*16;
            const uint32_t loff =
                (uint32_t)(uintptr_t)(x_raw + nxt*(16*256) + mm*256 + chunk*16);
            async_copy64(loff, ga);
        }

        // ---- init accumulators with biases ----
        v8f acc_rz[4]; v8f acc_in[2]; v8f acc_hn[2];
        #pragma unroll
        for (int i = 0; i < 4; ++i)
            #pragma unroll
            for (int v = 0; v < 8; ++v) acc_rz[i][v] = bias_rz[i];
        #pragma unroll
        for (int i = 0; i < 2; ++i)
            #pragma unroll
            for (int v = 0; v < 8; ++v) { acc_in[i][v] = bias_in[i]; acc_hn[i][v] = bias_hn[i]; }

        // ---- WMMA phase: both A-fragments come from LDS ----
        for (int k = 0; k < KTILES; ++k) {
            const v16bf ax = *(const v16bf*)(x_frag + (size_t)cur*4096 + (k*32 + lane)*16);
            const v16bf ah = *(const v16bf*)(h_frag + (k*32 + lane)*16);

            #pragma unroll
            for (int i = 0; i < 4; ++i) {
                const v16bf bih = *(const v16bf*)(packIH + ((size_t)((rz0 + i)*KTILES + k)*32 + lane)*16);
                acc_rz[i] = __builtin_amdgcn_wmma_f32_16x16x32_bf16(
                    false, ax, false, bih, (short)0, acc_rz[i], false, false);
                const v16bf bhh = *(const v16bf*)(packHH + ((size_t)((rz0 + i)*KTILES + k)*32 + lane)*16);
                acc_rz[i] = __builtin_amdgcn_wmma_f32_16x16x32_bf16(
                    false, ah, false, bhh, (short)0, acc_rz[i], false, false);
            }
            #pragma unroll
            for (int i = 0; i < 2; ++i) {
                const v16bf bih = *(const v16bf*)(packIH + ((size_t)((nn0 + i)*KTILES + k)*32 + lane)*16);
                acc_in[i] = __builtin_amdgcn_wmma_f32_16x16x32_bf16(
                    false, ax, false, bih, (short)0, acc_in[i], false, false);
                const v16bf bhh = *(const v16bf*)(packHH + ((size_t)((nn0 + i)*KTILES + k)*32 + lane)*16);
                acc_hn[i] = __builtin_amdgcn_wmma_f32_16x16x32_bf16(
                    false, ah, false, bhh, (short)0, acc_hn[i], false, false);
            }
        }

        // ---- spill gate pre-activations to LDS (C/D layout: M=v+8*hlf, N=m) ----
        #pragma unroll
        for (int i = 0; i < 4; ++i)
            #pragma unroll
            for (int v = 0; v < 8; ++v)
                g_rz[v + hlf*8][(rz0 + i)*16 + m] = acc_rz[i][v];
        #pragma unroll
        for (int i = 0; i < 2; ++i)
            #pragma unroll
            for (int v = 0; v < 8; ++v) {
                g_in[v + hlf*8][(nn0 + i - 32)*16 + m] = acc_in[i][v];
                g_hn[v + hlf*8][(nn0 + i - 32)*16 + m] = acc_hn[i][v];
            }

        wait_async0();          // staged x_{t+1} rows complete (this wave)
        __syncthreads();        // ... and visible to all waves

        // ---- elementwise GRU update + next-step fragment conversion ----
        {
            float hv[16];
            #pragma unroll
            for (int jj = 0; jj < 16; ++jj) {
                const int j = j0 + jj;
                const float r  = 1.f / (1.f + __expf(-g_rz[mm][j]));
                const float z  = 1.f / (1.f + __expf(-g_rz[mm][256 + j]));
                const float n  = tanhf(g_in[mm][j] + r * g_hn[mm][j]);
                const float hn = (1.f - z) * n + z * h_row[mm][j];
                h_row[mm][j] = hn;
                hv[jj] = hn;
            }
            pack_store_frag(hfrag_lo, hfrag_hi, hv);

            if (t + 1 < TT) {
                const float msk = ((start_mm + t + 1) <= end_mm) ? 1.f : 0.f;
                const float* xr = x_raw + nxt*(16*256) + mm*256 + j0;
                float xv[16];
                #pragma unroll
                for (int jj = 0; jj < 16; ++jj) xv[jj] = xr[jj] * msk;
                pack_store_frag(x_frag + (size_t)nxt*4096 + xf_lo,
                                x_frag + (size_t)nxt*4096 + xf_hi, xv);
            }
        }
        __syncthreads();
    }

    // ---- write final hidden state ----
    for (int i = tid; i < 16*256; i += 256) {
        const int r = i >> 8, d = i & 255;
        hidden[(size_t)(blockIdx.x*16 + r)*DD + d] = h_row[r][d];
    }
}

// ---------------------------------------------------------------------------
// Kernel 4: streaming saliency (bandwidth-bound, 256 MB of reads).
// out[b,t] = ((<hidden[b],src_vid[b,t]> + 1) * <memory[b,t], wcol> + bsum)/16
// One wave32 per (b,t); float4 loads; shuffle reduction.
// ---------------------------------------------------------------------------
__global__ __launch_bounds__(256) void saliency_kernel(
    const float* __restrict__ memory, const float* __restrict__ src_vid,
    const float* __restrict__ hidden, const float* __restrict__ wcol,
    const float* __restrict__ bsum, float* __restrict__ out)
{
    const int lane = threadIdx.x & 31;
    const int wave = threadIdx.x >> 5;
    const int pair = blockIdx.x * 8 + wave;      // pair = b*T + t, 0..131071
    const int b    = pair >> 11;

    const float4* sv = (const float4*)(src_vid + (size_t)pair * DD) + lane * 2;
    const float4* mv = (const float4*)(memory  + (size_t)pair * DD) + lane * 2;
    const float4* hv = (const float4*)(hidden  + (size_t)b    * DD) + lane * 2;
    const float4* wv = (const float4*)(wcol)                        + lane * 2;

    const float4 s0 = sv[0], s1 = sv[1], h0 = hv[0], h1 = hv[1];
    const float4 m0 = mv[0], m1 = mv[1], w0 = wv[0], w1 = wv[1];

    float wsum = s0.x*h0.x + s0.y*h0.y + s0.z*h0.z + s0.w*h0.w
               + s1.x*h1.x + s1.y*h1.y + s1.z*h1.z + s1.w*h1.w;
    float msum = m0.x*w0.x + m0.y*w0.y + m0.z*w0.z + m0.w*w0.w
               + m1.x*w1.x + m1.y*w1.y + m1.z*w1.z + m1.w*w1.w;

    #pragma unroll
    for (int off = 16; off > 0; off >>= 1) {
        wsum += __shfl_xor(wsum, off, 32);
        msum += __shfl_xor(msum, off, 32);
    }
    if (lane == 0)
        out[pair] = ((wsum + 1.f) * msum + *bsum) * 0.0625f;  // 1/sqrt(256)
}

// ---------------------------------------------------------------------------
// Host launcher
// ---------------------------------------------------------------------------
extern "C" void kernel_launch(void* const* d_in, const int* in_sizes, int n_in,
                              void* d_out, int out_size, void* d_ws, size_t ws_size,
                              hipStream_t stream)
{
    const float* pred_logits = (const float*)d_in[0];
    const float* pred_spans  = (const float*)d_in[1];
    const float* memory      = (const float*)d_in[2];
    const float* src_vid     = (const float*)d_in[3];
    const float* W_ih        = (const float*)d_in[4];
    const float* W_hh        = (const float*)d_in[5];
    const float* b_ih        = (const float*)d_in[6];
    const float* b_hh        = (const float*)d_in[7];
    const float* W_sal       = (const float*)d_in[8];
    const float* b_sal       = (const float*)d_in[9];
    float* out = (float*)d_out;

    // workspace layout (byte offsets)
    char* ws = (char*)d_ws;
    int*    ws_start  = (int*)(ws + 0);            //  64 ints   (256 B)
    int*    ws_end    = (int*)(ws + 256);          //  64 ints   (256 B)
    float*  ws_wcol   = (float*)(ws + 512);        // 256 floats (1 KB)
    float*  ws_bsum   = (float*)(ws + 1536);       //   1 float  (pad to 256 B)
    float*  ws_hidden = (float*)(ws + 1792);       // 64*256 f32 (64 KB)
    __bf16* ws_packIH = (__bf16*)(ws + 1792 + 65536);                 // 384 KB
    __bf16* ws_packHH = (__bf16*)(ws + 1792 + 65536 + 2*PACK_ELEMS);  // 384 KB

    prep_kernel<<<1, 256, 0, stream>>>(pred_logits, pred_spans, W_sal, b_sal,
                                       ws_start, ws_end, ws_wcol, ws_bsum);

    pack_kernel<<<(2*PACK_ELEMS)/256, 256, 0, stream>>>(W_ih, W_hh, ws_packIH, ws_packHH);

    // 32K(g_rz)+16K(g_in)+16K(g_hn)+16K(h_row)+8K(h_frag)+16K(x_frag)+32K(x_raw)
    const size_t gru_lds = 139264;
    gru_kernel<<<4, 256, gru_lds, stream>>>(src_vid, b_ih, b_hh,
                                            ws_start, ws_end,
                                            ws_packIH, ws_packHH, ws_hidden);

    saliency_kernel<<<(BQ*TT)/8, 256, 0, stream>>>(memory, src_vid, ws_hidden,
                                                   ws_wcol, ws_bsum, out);
}
